// GraphGPT_86011015070094
// MI455X (gfx1250) — matrix operand order repaired and checked by reading, not compiled, not run on hardware
//
#include <hip/hip_runtime.h>
#include <hip/hip_bf16.h>

// ---------------------------------------------------------------------------
// GraphGPT GCN fusion for MI455X (gfx1250, wave32, WMMA)
//   tokens = pool(rgb,ir)+pos ; 8x [ x@W -> gather/scatter -> relu ] ; LN
// GEMM uses V_WMMA_F32_16X16X32_BF16 with a bf16-hi/lo split of the fp32
// operands (3 WMMAs per K=32: hi*hi + lo*hi + hi*lo) for near-fp32 accuracy.
// ---------------------------------------------------------------------------

#define CDIM      512
#define NNODES    128
#define NEDGES    1024
#define NBATCH    64
#define NLAYERS   8
#define MROWS     (NBATCH * NNODES)   // 8192

typedef __attribute__((ext_vector_type(16))) __bf16 v16bf;
typedef __attribute__((ext_vector_type(8)))  __bf16 v8bf;
typedef __attribute__((ext_vector_type(8)))  float  v8f;

// round-to-nearest-even f32 -> bf16 bits
__device__ __forceinline__ unsigned short bf16_rn(float f) {
    union { float f; unsigned u; } v; v.f = f;
    unsigned r = v.u + 0x7FFFu + ((v.u >> 16) & 1u);
    return (unsigned short)(r >> 16);
}
__device__ __forceinline__ float bf16_to_f32(unsigned short h) {
    union { unsigned u; float f; } v; v.u = ((unsigned)h) << 16;
    return v.f;
}

// ---------------------------------------------------------------------------
// 1) 4x4 adaptive average pool + pos-embed -> x[(b*128+n)*512 + c]
//    thread = (b, group, c, spatial): fully coalesced float4 reads
// ---------------------------------------------------------------------------
__global__ __launch_bounds__(256) void pool_kernel(
    const float* __restrict__ rgb, const float* __restrict__ ir,
    const float* __restrict__ pos, float* __restrict__ x)
{
    int tid = blockIdx.x * 256 + threadIdx.x;       // 4,194,304 total
    int s = tid & 63;                               // va*8+ha
    int c = (tid >> 6) & 511;
    int g = (tid >> 15) & 1;
    int b = tid >> 16;
    int va = s >> 3, ha = s & 7;

    const float* src = (g ? ir : rgb)
        + ((size_t)(b * CDIM + c) * 32 + va * 4) * 32 + ha * 4;
    float sum = 0.0f;
#pragma unroll
    for (int dy = 0; dy < 4; ++dy) {
        float4 v = *(const float4*)(src + dy * 32);
        sum += v.x + v.y + v.z + v.w;
    }
    int n = g * 64 + s;
    x[(size_t)(b * NNODES + n) * CDIM + c] = pos[n * CDIM + c] + sum * (1.0f / 16.0f);
}

// ---------------------------------------------------------------------------
// 2) build CSR over dst + symmetric GCN norm (single block, LDS)
// ---------------------------------------------------------------------------
__global__ __launch_bounds__(256) void graph_build_kernel(
    const int* __restrict__ ei,      // (2, E): [0..E) = src, [E..2E) = dst
    int* __restrict__ row_ptr, int* __restrict__ col, float* __restrict__ nrm)
{
    __shared__ int   deg[NNODES];
    __shared__ int   fill[NNODES];
    __shared__ float dinv[NNODES];
    __shared__ int   rp[NNODES + 1];
    int t = threadIdx.x;
    if (t < NNODES) { deg[t] = 0; fill[t] = 0; }
    __syncthreads();
    for (int e = t; e < NEDGES + NNODES; e += 256) {
        int dst = (e < NEDGES) ? (ei[NEDGES + e] & (NNODES - 1)) : (e - NEDGES);
        atomicAdd(&deg[dst], 1);
    }
    __syncthreads();
    if (t < NNODES) dinv[t] = rsqrtf((float)deg[t]);   // deg >= 1 (self loop)
    if (t == 0) {
        rp[0] = 0;
        for (int n = 0; n < NNODES; ++n) rp[n + 1] = rp[n] + deg[n];
    }
    __syncthreads();
    for (int e = t; e < NEDGES + NNODES; e += 256) {
        int src = (e < NEDGES) ? (ei[e] & (NNODES - 1)) : (e - NEDGES);
        int dst = (e < NEDGES) ? (ei[NEDGES + e] & (NNODES - 1)) : (e - NEDGES);
        int pos = rp[dst] + atomicAdd(&fill[dst], 1);
        col[pos] = src;
        nrm[pos] = dinv[src] * dinv[dst];
    }
    if (t <= NNODES) row_ptr[t] = rp[t];
}

// ---------------------------------------------------------------------------
// 3) split fp32 -> bf16 hi/lo (x, same layout)
// ---------------------------------------------------------------------------
__global__ __launch_bounds__(256) void xsplit_kernel(
    const float* __restrict__ x,
    unsigned short* __restrict__ xh, unsigned short* __restrict__ xl)
{
    int tid = blockIdx.x * 256 + threadIdx.x;       // MROWS*CDIM
    float f = x[tid];
    unsigned short h = bf16_rn(f);
    xh[tid] = h;
    xl[tid] = bf16_rn(f - bf16_to_f32(h));
}

// split all layer weights and transpose: whT[l][e*512 + d] = bf16(W[l][d][e])
__global__ __launch_bounds__(256) void wsplit_kernel(
    const float* __restrict__ W,
    unsigned short* __restrict__ whT, unsigned short* __restrict__ wlT)
{
    int tid = blockIdx.x * 256 + threadIdx.x;       // 8*512*512
    int l = tid >> 18;
    int r = tid & 262143;
    int d = r >> 9;
    int e = r & 511;
    float f = W[tid];
    unsigned short h = bf16_rn(f);
    size_t o = (size_t)l * CDIM * CDIM + (size_t)e * CDIM + d;
    whT[o] = h;
    wlT[o] = bf16_rn(f - bf16_to_f32(h));
}

// ---------------------------------------------------------------------------
// 4) GEMM: xw[m, e] = sum_d x[m, d] * W[d, e]   (M=8192, N=512, K=512)
//    bf16x3 split, V_WMMA_F32_16X16X32_BF16, wave32.
//    Each wave: one 16-row M tile x four 16-col N tiles (A reused 4x).
//    A layout (16-bit 16x32): lane<16 -> K 0..15,        lane>=16 -> K 16..31
//    B layout (32x16):        lane<16 -> K 0..15,        lane>=16 -> K 16..31
// ---------------------------------------------------------------------------
__global__ __launch_bounds__(256) void gemm_bf16x3_kernel(
    const __bf16* __restrict__ xh, const __bf16* __restrict__ xl,
    const __bf16* __restrict__ whT, const __bf16* __restrict__ wlT,
    float* __restrict__ xw)
{
    const int lane = threadIdx.x & 31;
    const int wid  = blockIdx.x * 8 + (threadIdx.x >> 5);   // 4096 wave jobs
    const int mt = wid >> 3;          // 512 M tiles
    const int ng = wid & 7;           // 8 groups of 4 N tiles
    const int m0 = mt * 16;
    const int n0 = ng * 64;

    const int arow = m0 + (lane & 15);
    const int kA = (lane >> 4) * 16;  // A half-wave K base
    const int kB = (lane >> 4) * 16;  // B half-wave K base

    v8f acc[4] = {v8f{0,0,0,0,0,0,0,0}, v8f{0,0,0,0,0,0,0,0},
                  v8f{0,0,0,0,0,0,0,0}, v8f{0,0,0,0,0,0,0,0}};

    union AU { v16bf v; v8bf h[2]; };

    for (int k0 = 0; k0 < CDIM; k0 += 32) {
        AU ah, al;
        const __bf16* pah = xh + (size_t)arow * CDIM + k0 + kA;
        const __bf16* pal = xl + (size_t)arow * CDIM + k0 + kA;
        ah.h[0] = *(const v8bf*)(pah);
        ah.h[1] = *(const v8bf*)(pah + 8);
        al.h[0] = *(const v8bf*)(pal);
        al.h[1] = *(const v8bf*)(pal + 8);

#pragma unroll
        for (int j = 0; j < 4; ++j) {
            int coln = n0 + j * 16 + (lane & 15);
            v16bf bh = *(const v16bf*)(whT + (size_t)coln * CDIM + k0 + kB);
            v16bf bl = *(const v16bf*)(wlT + (size_t)coln * CDIM + k0 + kB);
            acc[j] = __builtin_amdgcn_wmma_f32_16x16x32_bf16(
                false, ah.v, false, bh, (short)0, acc[j], false, false);
            acc[j] = __builtin_amdgcn_wmma_f32_16x16x32_bf16(
                false, al.v, false, bh, (short)0, acc[j], false, false);
            acc[j] = __builtin_amdgcn_wmma_f32_16x16x32_bf16(
                false, ah.v, false, bl, (short)0, acc[j], false, false);
        }
    }

    // C/D layout: VGPR i -> M = m0 + (lane>>4)*8 + i, N = n0 + j*16 + (lane&15)
    const int crow0 = m0 + (lane >> 4) * 8;
    const int ccol  = lane & 15;
#pragma unroll
    for (int j = 0; j < 4; ++j) {
        float* op = xw + (size_t)crow0 * CDIM + n0 + j * 16 + ccol;
#pragma unroll
        for (int i = 0; i < 8; ++i) op[(size_t)i * CDIM] = acc[j][i];
    }
}

// ---------------------------------------------------------------------------
// 5) CSR aggregate + bias + relu -> new x (atomic-free scatter)
// ---------------------------------------------------------------------------
__global__ __launch_bounds__(256) void aggregate_kernel(
    const float* __restrict__ xw, const float* __restrict__ bias,
    const int* __restrict__ row_ptr, const int* __restrict__ col,
    const float* __restrict__ nrm, float* __restrict__ x)
{
    int tid = blockIdx.x * 256 + threadIdx.x;   // MROWS*CDIM
    int c = tid & 511;
    int m = tid >> 9;
    int n = m & (NNODES - 1);
    int b = m >> 7;
    float s = bias[c];
    int p0 = row_ptr[n], p1 = row_ptr[n + 1];
    const float* xwb = xw + (size_t)b * NNODES * CDIM;
    for (int p = p0; p < p1; ++p)
        s += xwb[(size_t)col[p] * CDIM + c] * nrm[p];
    x[(size_t)m * CDIM + c] = fmaxf(s, 0.0f);
}

// ---------------------------------------------------------------------------
// 6) LayerNorm over C + transpose to (2,B,C,8,8), split rgb/ir halves
// ---------------------------------------------------------------------------
__global__ __launch_bounds__(256) void ln_out_kernel(
    const float* __restrict__ x, const float* __restrict__ lnw,
    const float* __restrict__ lnb, float* __restrict__ out)
{
    __shared__ float s1[256];
    __shared__ float s2[256];
    int token = blockIdx.x;             // 8192 tokens
    int b = token >> 7;
    int n = token & (NNODES - 1);
    int t = threadIdx.x;
    const float* xp = x + (size_t)token * CDIM;
    float v0 = xp[t], v1 = xp[t + 256];
    s1[t] = v0 + v1;
    s2[t] = v0 * v0 + v1 * v1;
    __syncthreads();
    for (int off = 128; off > 0; off >>= 1) {
        if (t < off) { s1[t] += s1[t + off]; s2[t] += s2[t + off]; }
        __syncthreads();
    }
    float mu  = s1[0] * (1.0f / CDIM);
    float var = s2[0] * (1.0f / CDIM) - mu * mu;
    float rstd = rsqrtf(var + 1e-5f);

    int g = n >> 6;       // 0 = rgb, 1 = ir
    int s = n & 63;       // spatial va*8+ha
    float* ob = out + (size_t)g * (NBATCH * CDIM * 64)
                    + (size_t)b * CDIM * 64 + s;
    ob[(size_t)t * 64]         = (v0 - mu) * rstd * lnw[t]       + lnb[t];
    ob[(size_t)(t + 256) * 64] = (v1 - mu) * rstd * lnw[t + 256] + lnb[t + 256];
}

// ---------------------------------------------------------------------------
// launcher
// ---------------------------------------------------------------------------
extern "C" void kernel_launch(void* const* d_in, const int* in_sizes, int n_in,
                              void* d_out, int out_size, void* d_ws, size_t ws_size,
                              hipStream_t stream)
{
    const float* rgb = (const float*)d_in[0];
    const float* ir  = (const float*)d_in[1];
    const int*   ei  = (const int*)  d_in[2];
    const float* pos = (const float*)d_in[3];
    const float* Wg  = (const float*)d_in[4];
    const float* bg  = (const float*)d_in[5];
    const float* lnw = (const float*)d_in[6];
    const float* lnb = (const float*)d_in[7];
    float* out = (float*)d_out;

    char* ws = (char*)d_ws;
    float*          x    = (float*)(ws);                               // 16 MB
    float*          xw   = (float*)(ws + (size_t)(16u << 20));         // 16 MB
    unsigned short* xh   = (unsigned short*)(ws + (size_t)(32u << 20));//  8 MB
    unsigned short* xl   = (unsigned short*)(ws + (size_t)(40u << 20));//  8 MB
    unsigned short* whT  = (unsigned short*)(ws + (size_t)(48u << 20));//  4 MB
    unsigned short* wlT  = (unsigned short*)(ws + (size_t)(52u << 20));//  4 MB
    int*            rptr = (int*)(ws + (size_t)(56u << 20));
    int*            colb = rptr + 256;
    float*          nrmb = (float*)(colb + 2048);

    // split+transpose all 8 layer weights (2,097,152 elems)
    wsplit_kernel<<<8192, 256, 0, stream>>>(Wg, whT, wlT);
    // pooled tokens + positional embedding (4,194,304 elems)
    pool_kernel<<<16384, 256, 0, stream>>>(rgb, ir, pos, x);
    // CSR + gcn norm
    graph_build_kernel<<<1, 256, 0, stream>>>(ei, rptr, colb, nrmb);

    for (int l = 0; l < NLAYERS; ++l) {
        xsplit_kernel<<<16384, 256, 0, stream>>>(x, xh, xl);
        gemm_bf16x3_kernel<<<512, 256, 0, stream>>>(
            (const __bf16*)xh, (const __bf16*)xl,
            (const __bf16*)(whT + (size_t)l * CDIM * CDIM),
            (const __bf16*)(wlT + (size_t)l * CDIM * CDIM), xw);
        aggregate_kernel<<<16384, 256, 0, stream>>>(
            xw, bg + l * CDIM, rptr, colb, nrmb, x);
    }

    ln_out_kernel<<<MROWS, 256, 0, stream>>>(x, lnw, lnb, out);
}
